// QAConv_43104291783358
// MI455X (gfx1250) — matrix operand order repaired and checked by reading, not compiled, not run on hardware
//
#include <hip/hip_runtime.h>

// ---------------------------------------------------------------------------
// QAConv fused kernel for MI455X (gfx1250, wave32, WMMA).
// s[b,g,i,j] = sum_c gal_n[g,c,i] * prob_n[b,c,j]  (192x192x512 GEMM per pair)
// fused with dual max-pool + scalar BN + fc dot + logit BN + sigmoid.
// Double-buffered LDS pipeline: global loads for chunk k+1 overlap the 12
// WMMAs of chunk k; one barrier per K-chunk.
// ---------------------------------------------------------------------------

#define C_DIM   512
#define HW      192
#define B_DIM   16
#define G_DIM   128
#define KCHUNKS (C_DIM / 32)
#define LDSTR   40            // bf16 elements per LDS row (32 data + 8 pad)

typedef __attribute__((ext_vector_type(16))) __bf16 v16bf;
typedef __attribute__((ext_vector_type(8)))  float  v8f;

__device__ __forceinline__ v16bf load_frag16(const __bf16* p0, const __bf16* p1) {
    union { uint4 u[2]; v16bf v; } r;
    r.u[0] = *(const uint4*)p0;   // ds_load_b128
    r.u[1] = *(const uint4*)p1;   // ds_load_b128
    return r.v;
}

// ---------------------------------------------------------------------------
// Pass 1: channel-L2 normalize + transpose to [N, hw, C] bf16 (K-contiguous,
// exactly the per-lane layout WMMA A/B fragments want). One wave per row.
// ---------------------------------------------------------------------------
__global__ void l2norm_transpose_kernel(const float* __restrict__ x,
                                        __bf16* __restrict__ out) {
    const int row  = blockIdx.x * (blockDim.x >> 5) + (threadIdx.x >> 5);
    const int lane = threadIdx.x & 31;
    const int n = row / HW;
    const int j = row % HW;
    const float* base = x + (size_t)n * C_DIM * HW + j;

    float ss = 0.f;
    #pragma unroll
    for (int c = lane; c < C_DIM; c += 32) {
        float v = base[(size_t)c * HW];
        ss += v * v;
    }
    #pragma unroll
    for (int m = 16; m >= 1; m >>= 1) ss += __shfl_xor(ss, m, 32);

    const float inv = 1.0f / fmaxf(sqrtf(ss), 1e-12f);
    __bf16* o = out + (size_t)row * C_DIM;
    #pragma unroll
    for (int c = lane; c < C_DIM; c += 32)
        o[c] = (__bf16)(base[(size_t)c * HW] * inv);
}

// ---------------------------------------------------------------------------
// Pass 2: one workgroup per (b,g) pair. 12 waves; wave w owns i-tile w and
// keeps 12 f32 accumulators (one per j-tile). K staged through double-
// buffered LDS in 192x32 bf16 slices.
// ---------------------------------------------------------------------------
__global__ void __launch_bounds__(384, 1)
qaconv_wmma_kernel(const __bf16* __restrict__ galT,
                   const __bf16* __restrict__ probT,
                   const float* __restrict__ bn_w,  const float* __restrict__ bn_b,
                   const float* __restrict__ bn_m,  const float* __restrict__ bn_v,
                   const float* __restrict__ fcw,   const float* __restrict__ fcb,
                   const float* __restrict__ lbn_w, const float* __restrict__ lbn_b,
                   const float* __restrict__ lbn_m, const float* __restrict__ lbn_v,
                   float* __restrict__ out) {
    __shared__ __align__(16) __bf16 Atile[2][HW * LDSTR];   // gal rows (i, K)
    __shared__ __align__(16) __bf16 Btile[2][HW * LDSTR];   // prob rows (j, K)
    // Epilogue arrays alias the tile storage: only touched after the last
    // tile read, and the K-loop ends on a barrier. 11.5KB << 30KB buffer.
    float* const colpart = (float*)&Atile[0][0];            // [12][HW]
    float* const rowarr  = colpart + 12 * HW;               // [HW]
    float* const red     = rowarr + HW;                     // [384]

    const int tid  = threadIdx.x;
    const int lane = tid & 31;
    const int wave = tid >> 5;        // 0..11  == i-tile index
    const int half = lane >> 4;       // 0/1    == K-half selector
    const int mn   = lane & 15;       // M (A) / N (B,D) within tile
    const int bid  = blockIdx.x;
    const int b = bid / G_DIM, g = bid % G_DIM;

    const __bf16* gbase = galT  + (size_t)g * HW * C_DIM;
    const __bf16* pbase = probT + (size_t)b * HW * C_DIM;

    // Staging: 768 16B chunks per matrix per K-chunk; thread t covers rows
    // (t>>2) and (t>>2)+96 at 16B sub-chunk (t&3).
    const int    srow = tid >> 2;
    const int    ssub = tid & 3;
    const size_t goff = (size_t)srow * C_DIM + ssub * 8;
    const int    loff = srow * LDSTR + ssub * 8;

    uint4 ar0, ar1, br0, br1;
    auto stage = [&](int kc) {
        const size_t o0 = goff + kc * 32;
        const size_t o1 = o0 + (size_t)96 * C_DIM;
        ar0 = *(const uint4*)(gbase + o0);
        ar1 = *(const uint4*)(gbase + o1);
        br0 = *(const uint4*)(pbase + o0);
        br1 = *(const uint4*)(pbase + o1);
    };
    auto commit = [&](int buf) {
        *(uint4*)(&Atile[buf][loff])             = ar0;
        *(uint4*)(&Atile[buf][loff + 96 * LDSTR]) = ar1;
        *(uint4*)(&Btile[buf][loff])             = br0;
        *(uint4*)(&Btile[buf][loff + 96 * LDSTR]) = br1;
    };

    stage(0);
    commit(0);
    __syncthreads();

    v8f acc[12] = {};                 // 12 j-tile accumulators (96 VGPRs)

    #pragma unroll
    for (int kc = 0; kc < KCHUNKS; ++kc) {
        const int cur = kc & 1;
        if (kc + 1 < KCHUNKS) stage(kc + 1);   // overlap with WMMA below

        // A fragment: 16x32 bf16. lane half selects K-offset {0,8}; lane
        // holds K = {koff..koff+7, koff+16..koff+23} of row i = 16*wave + mn.
        const __bf16* arow = &Atile[cur][(wave * 16 + mn) * LDSTR + half * 8];
        const v16bf afrag = load_frag16(arow, arow + 16);

        #pragma unroll
        for (int t = 0; t < 12; ++t) {
            // B fragment: 32x16 bf16. lane holds K = {koff..koff+7, koff+16..koff+23}
            // of column j = 16*t + mn, koff = 8*half.
            const __bf16* brow = &Btile[cur][(t * 16 + mn) * LDSTR + half * 8];
            const v16bf bfrag = load_frag16(brow, brow + 16);
            acc[t] = __builtin_amdgcn_wmma_f32_16x16x32_bf16(
                         false, afrag, false, bfrag,
                         (short)0, acc[t], false, false);
        }

        if (kc + 1 < KCHUNKS) commit(cur ^ 1);
        __syncthreads();   // RAW for next compute; WAR for the store above
    }

    // ---- Fused dual max-pool (loop above ended on a barrier) ------------
    // D layout: VGPR r, lane<16 -> S[M=r][N=lane]; lane>=16 -> S[M=r+8][N=lane-16]

    // Column max (max over i) -> one value per probe position j.
    #pragma unroll
    for (int t = 0; t < 12; ++t) {
        float cm = acc[t][0];
        #pragma unroll
        for (int r = 1; r < 8; ++r) cm = fmaxf(cm, acc[t][r]);
        cm = fmaxf(cm, __shfl_xor(cm, 16, 32));        // merge M halves
        if (half == 0) colpart[wave * HW + t * 16 + mn] = cm;
    }
    // Row max (max over j) -> one value per gallery position i.
    #pragma unroll
    for (int r = 0; r < 8; ++r) {
        float rm = acc[0][r];
        #pragma unroll
        for (int t = 1; t < 12; ++t) rm = fmaxf(rm, acc[t][r]);
        #pragma unroll
        for (int m = 8; m >= 1; m >>= 1) rm = fmaxf(rm, __shfl_xor(rm, m, 32));
        if (mn == 0) rowarr[wave * 16 + half * 8 + r] = rm;
    }
    __syncthreads();

    // ---- BN affine + fc dot + logit BN + sigmoid -----------------------
    const float bscale = bn_w[0] * rsqrtf(bn_v[0] + 1e-5f);
    const float bmean  = bn_m[0];
    const float bbias  = bn_b[0];

    float val;
    if (tid < HW) {                    // score[0:hw] = max over i (per j)
        float m = colpart[tid];
        #pragma unroll
        for (int w = 1; w < 12; ++w) m = fmaxf(m, colpart[w * HW + tid]);
        val = fcw[tid] * ((m - bmean) * bscale + bbias);
    } else {                           // score[hw:2hw] = max over j (per i)
        float m = rowarr[tid - HW];
        val = fcw[tid] * ((m - bmean) * bscale + bbias);
    }
    red[tid] = val;
    __syncthreads();
    if (tid < 128) red[tid] += red[tid + 128] + red[tid + 256];
    __syncthreads();
    if (tid < 64)  red[tid] += red[tid + 64];
    __syncthreads();
    if (tid < 32) {
        float v = red[tid] + red[tid + 32];
        #pragma unroll
        for (int m = 16; m >= 1; m >>= 1) v += __shfl_xor(v, m, 32);
        if (tid == 0) {
            const float logits = v + fcb[0];
            const float l = (logits - lbn_m[0]) *
                            (lbn_w[0] * rsqrtf(lbn_v[0] + 1e-5f)) + lbn_b[0];
            out[bid] = 1.0f / (1.0f + expf(-l * 0.1f));
        }
    }
}

// ---------------------------------------------------------------------------
extern "C" void kernel_launch(void* const* d_in, const int* in_sizes, int n_in,
                              void* d_out, int out_size, void* d_ws, size_t ws_size,
                              hipStream_t stream) {
    const float* prob = (const float*)d_in[0];   // [16,512,24,8]
    const float* gal  = (const float*)d_in[1];   // [128,512,24,8]

    __bf16* galT  = (__bf16*)d_ws;                                // [128,192,512]
    __bf16* probT = galT + (size_t)G_DIM * HW * C_DIM;            // [16,192,512]

    // 8 waves per block in the normalize pass.
    l2norm_transpose_kernel<<<(G_DIM * HW) / 8, 256, 0, stream>>>(gal, galT);
    l2norm_transpose_kernel<<<(B_DIM * HW) / 8, 256, 0, stream>>>(prob, probT);

    qaconv_wmma_kernel<<<B_DIM * G_DIM, 384, 0, stream>>>(
        galT, probT,
        (const float*)d_in[2],  (const float*)d_in[3],
        (const float*)d_in[4],  (const float*)d_in[5],
        (const float*)d_in[6],  (const float*)d_in[7],
        (const float*)d_in[8],  (const float*)d_in[9],
        (const float*)d_in[10], (const float*)d_in[11],
        (float*)d_out);
}
